// PipelinedMoEBlock_50680614092863
// MI455X (gfx1250) — compile-verified, matrix operand order checked by reading, not run on hardware
//
#include <hip/hip_runtime.h>
#include <math.h>

// ---------------- problem constants (match reference) ----------------
#define N_TOK 4096      // B*T
#define DIMD  1024
#define FFD   4096
#define NEXP  8
#define NTOPK 2
#define SLOTS (N_TOK * NTOPK)   // 8192 routed (token,expert) pairs

typedef __attribute__((ext_vector_type(16))) __bf16 v16bf;
typedef __attribute__((ext_vector_type(8)))  float  v8f;

// ---------------- WMMA fragment gather from LDS ----------------
// LDS tile layout: [rows][32 K] bf16, row stride 32 elements (64B).
// 16-bit A/B layout (ISA 7.12.2): lane<16 holds K{0..7,16..23},
// lane>=16 holds K{8..15,24..31}; row = tile_base + (lane&15).
union FragU { v16bf v; float4 f4[2]; };

__device__ __forceinline__ v16bf load_frag(const __bf16* lds, int row, int khalf) {
    const __bf16* p = lds + row * 32;
    FragU u;
    u.f4[0] = *reinterpret_cast<const float4*>(p + 8 * khalf);        // K run 0
    u.f4[1] = *reinterpret_cast<const float4*>(p + 16 + 8 * khalf);   // K run 1
    return u.v;
}

// memory -> LDS direct async copy (ASYNCcnt), 16B per lane
__device__ __forceinline__ void async_copy_b128(unsigned lds_byte_off, const __bf16* gsrc) {
    asm volatile("global_load_async_to_lds_b128 %0, %1, off"
                 :: "v"(lds_byte_off), "v"(gsrc) : "memory");
}
__device__ __forceinline__ unsigned lds_off(const __bf16* p) {
    return (unsigned)(uintptr_t)p;   // generic shared VA low 32 bits = LDS offset
}

__device__ __forceinline__ float gelu_exact(float v) {
    return 0.5f * v * (1.0f + erff(v * 0.70710678118654752440f));
}

// ---------------- 1) x fp32 -> bf16 ----------------
__global__ void k_cvt_x(const float* __restrict__ x, __bf16* __restrict__ xb) {
    int i = (blockIdx.x * 256 + threadIdx.x) * 4;
    float4 v = *reinterpret_cast<const float4*>(x + i);
    xb[i + 0] = (__bf16)v.x;
    xb[i + 1] = (__bf16)v.y;
    xb[i + 2] = (__bf16)v.z;
    xb[i + 3] = (__bf16)v.w;
}

// ---------------- 2) per-expert transpose + convert: in[R][C] f32 -> out[C][R] bf16 ----------------
__global__ void k_transpose_cvt(const float* __restrict__ in, __bf16* __restrict__ out,
                                int R, int C) {
    int e = blockIdx.z;
    const float* src = in  + (size_t)e * R * C;
    __bf16*      dst = out + (size_t)e * R * C;
    __shared__ __bf16 tile[64][65];   // +1 pad: conflict-free transpose
    int r0 = blockIdx.y * 64, c0 = blockIdx.x * 64;
    for (int i = threadIdx.x; i < 64 * 64; i += 256) {
        int r = i >> 6, c = i & 63;                     // coalesced read along C
        tile[r][c] = (__bf16)src[(size_t)(r0 + r) * C + (c0 + c)];
    }
    __syncthreads();
    for (int i = threadIdx.x; i < 64 * 64; i += 256) {
        int c = i >> 6, r = i & 63;                     // coalesced write along R
        dst[(size_t)(c0 + c) * R + (r0 + r)] = tile[r][c];
    }
}

// ---------------- 3) router: logits, top-2, softmax ----------------
__global__ void k_router(const float* __restrict__ x, const float* __restrict__ Wr,
                         int* __restrict__ topi, float* __restrict__ topw) {
    int wave = threadIdx.x >> 5;
    int lane = threadIdx.x & 31;
    int n = blockIdx.x * 4 + wave;
    const float* xr = x + (size_t)n * DIMD;
    float acc[NEXP];
#pragma unroll
    for (int e = 0; e < NEXP; ++e) acc[e] = 0.0f;
    for (int d = lane; d < DIMD; d += 32) {
        float xv = xr[d];
#pragma unroll
        for (int e = 0; e < NEXP; ++e) acc[e] += xv * Wr[e * DIMD + d];
    }
#pragma unroll
    for (int e = 0; e < NEXP; ++e)
        for (int off = 16; off > 0; off >>= 1)
            acc[e] += __shfl_xor(acc[e], off, 32);
    if (lane == 0) {
        int i0 = 0; float v0 = acc[0];
#pragma unroll
        for (int e = 1; e < NEXP; ++e) if (acc[e] > v0) { v0 = acc[e]; i0 = e; }
        int i1 = -1; float v1 = -3.4e38f;
#pragma unroll
        for (int e = 0; e < NEXP; ++e) if (e != i0 && acc[e] > v1) { v1 = acc[e]; i1 = e; }
        float t = expf(v1 - v0);              // softmax over {v0,v1}, m = v0
        float inv = 1.0f / (1.0f + t);
        topi[n * 2 + 0] = i0;  topw[n * 2 + 0] = inv;
        topi[n * 2 + 1] = i1;  topw[n * 2 + 1] = t * inv;
    }
}

// ---------------- 4) compaction: counts, offsets, slot lists ----------------
__global__ void k_scatter(const int* __restrict__ topi, const float* __restrict__ topw,
                          int* __restrict__ tok, float* __restrict__ wgt,
                          int* __restrict__ counts, int* __restrict__ offsets) {
    __shared__ int cnt[NEXP], cur[NEXP], offs[NEXP + 1];
    int t = threadIdx.x;
    if (t < NEXP) { cnt[t] = 0; cur[t] = 0; }
    __syncthreads();
    for (int i = t; i < SLOTS; i += 256) atomicAdd(&cnt[topi[i]], 1);
    __syncthreads();
    if (t == 0) {
        offs[0] = 0;
        for (int e = 0; e < NEXP; ++e) offs[e + 1] = offs[e] + cnt[e];
    }
    __syncthreads();
    if (t < NEXP) { counts[t] = cnt[t]; offsets[t] = offs[t]; }
    if (t == 0) offsets[NEXP] = offs[NEXP];
    for (int i = t; i < SLOTS; i += 256) {
        int e = topi[i];
        int p = offs[e] + atomicAdd(&cur[e], 1);
        tok[p] = i >> 1;
        wgt[p] = topw[i];
    }
}

// ================= routed GEMM cores: 64(M) x 128(N) tile, double-buffered =================
// block = 256 threads (8 waves, 2x4 wave grid); each wave owns a 32x32 C block
// -> 4 v_wmma per wave per 32-K step from 2 A + 2 B fragment loads.
// LDS staged by GLOBAL_LOAD_ASYNC_TO_LDS_B128 (ASYNCcnt), double buffered so the
// copy of K-slab i+1 overlaps the WMMAs of slab i.

#define A_ELEMS (64 * 32)      // per buffer
#define B_ELEMS (128 * 32)     // per buffer

// ---------------- 5) GEMM1: h[slot][FF] = gelu(x[tok] @ W1[e]), bf16 out ----------------
// grid: (N_TOK/64 capacity M-tiles, FFD/128 N-tiles, NEXP)
__global__ void k_ffn1(const __bf16* __restrict__ xb, const __bf16* __restrict__ w1t,
                       const int* __restrict__ tok,
                       const int* __restrict__ counts, const int* __restrict__ offsets,
                       __bf16* __restrict__ h) {
    int e = blockIdx.z;
    int cnt = counts[e];
    int m0 = blockIdx.x * 64;
    if (m0 >= cnt) return;
    int base = offsets[e];
    int f0 = blockIdx.y * 128;
    const __bf16* w1e = w1t + (size_t)e * FFD * DIMD;   // [f][d] layout

    __shared__ __align__(16) __bf16 As[2 * A_ELEMS];
    __shared__ __align__(16) __bf16 Bs[2 * B_ELEMS];

    int t = threadIdx.x;
    int ldr = t >> 2;            // 0..63 staging row
    int ldc = (t & 3) * 8;       // 0,8,16,24 bf16 (16B)
    int srow = m0 + ldr; if (srow >= cnt) srow = cnt - 1;         // clamp (masked at store)
    const __bf16* asrc  = xb + (size_t)tok[base + srow] * DIMD + ldc;
    const __bf16* bsrc0 = w1e + (size_t)(f0 + ldr) * DIMD + ldc;
    const __bf16* bsrc1 = w1e + (size_t)(f0 + 64 + ldr) * DIMD + ldc;
    unsigned aL  = lds_off(As + ldr * 32 + ldc);
    unsigned bL0 = lds_off(Bs + ldr * 32 + ldc);
    unsigned bL1 = lds_off(Bs + (64 + ldr) * 32 + ldc);

    int wv = t >> 5, lane = t & 31;
    int wm = wv >> 2, wn = wv & 3;       // 2x4 wave grid
    int rit = lane & 15, khalf = lane >> 4;
    v8f acc00 = {}, acc01 = {}, acc10 = {}, acc11 = {};

    const int NSTEP = DIMD / 32;
    // prologue: stage 0 into buffer 0
    async_copy_b128(aL, asrc);
    async_copy_b128(bL0, bsrc0);
    async_copy_b128(bL1, bsrc1);

    for (int i = 0; i < NSTEP; ++i) {
        if (i + 1 < NSTEP) {
            int kn = (i + 1) * 32;
            unsigned bo = ((i + 1) & 1) ? 1u : 0u;
            async_copy_b128(aL  + bo * (A_ELEMS * 2), asrc  + kn);
            async_copy_b128(bL0 + bo * (B_ELEMS * 2), bsrc0 + kn);
            async_copy_b128(bL1 + bo * (B_ELEMS * 2), bsrc1 + kn);
            __builtin_prefetch(bsrc0 + kn + 32, 0, 1);
            asm volatile("s_wait_asynccnt 0x3" ::: "memory");   // stage i done (in-order)
        } else {
            asm volatile("s_wait_asynccnt 0x0" ::: "memory");
        }
        __syncthreads();
        const __bf16* Ab = As + (i & 1) * A_ELEMS;
        const __bf16* Bb = Bs + (i & 1) * B_ELEMS;
        v16bf a0 = load_frag(Ab, wm * 32 + rit,      khalf);
        v16bf a1 = load_frag(Ab, wm * 32 + 16 + rit, khalf);
        v16bf b0 = load_frag(Bb, wn * 32 + rit,      khalf);
        v16bf b1 = load_frag(Bb, wn * 32 + 16 + rit, khalf);
        acc00 = __builtin_amdgcn_wmma_f32_16x16x32_bf16(false, a0, false, b0, (short)0, acc00, false, false);
        acc01 = __builtin_amdgcn_wmma_f32_16x16x32_bf16(false, a0, false, b1, (short)0, acc01, false, false);
        acc10 = __builtin_amdgcn_wmma_f32_16x16x32_bf16(false, a1, false, b0, (short)0, acc10, false, false);
        acc11 = __builtin_amdgcn_wmma_f32_16x16x32_bf16(false, a1, false, b1, (short)0, acc11, false, false);
        __syncthreads();
    }

    int n_in = lane & 15;
    int m_hi = (lane >> 4) * 8;
    int gc0 = f0 + wn * 32 + n_in;
#pragma unroll
    for (int r = 0; r < 8; ++r) {
        int ml0 = wm * 32 + m_hi + r;        // C layout: m = r + 8*(lane>=16)
        int ml1 = ml0 + 16;
        if (m0 + ml0 < cnt) {
            size_t hrow = (size_t)(base + m0 + ml0) * FFD;
            h[hrow + gc0]      = (__bf16)gelu_exact(acc00[r]);
            h[hrow + gc0 + 16] = (__bf16)gelu_exact(acc01[r]);
        }
        if (m0 + ml1 < cnt) {
            size_t hrow = (size_t)(base + m0 + ml1) * FFD;
            h[hrow + gc0]      = (__bf16)gelu_exact(acc10[r]);
            h[hrow + gc0 + 16] = (__bf16)gelu_exact(acc11[r]);
        }
    }
}

// ---------------- 6) GEMM2: out[tok] += w * (h[slot] @ W2[e]) ----------------
// grid: (N_TOK/64, DIMD/128, NEXP)
__global__ void k_ffn2(const __bf16* __restrict__ h, const __bf16* __restrict__ w2t,
                       const int* __restrict__ tok, const float* __restrict__ wgt,
                       const int* __restrict__ counts, const int* __restrict__ offsets,
                       float* __restrict__ out) {
    int e = blockIdx.z;
    int cnt = counts[e];
    int m0 = blockIdx.x * 64;
    if (m0 >= cnt) return;
    int base = offsets[e];
    int d0 = blockIdx.y * 128;
    const __bf16* w2e = w2t + (size_t)e * DIMD * FFD;   // [d][f] layout

    __shared__ __align__(16) __bf16 As[2 * A_ELEMS];
    __shared__ __align__(16) __bf16 Bs[2 * B_ELEMS];

    int t = threadIdx.x;
    int ldr = t >> 2;
    int ldc = (t & 3) * 8;
    int srow = m0 + ldr; if (srow >= cnt) srow = cnt - 1;
    const __bf16* asrc  = h + (size_t)(base + srow) * FFD + ldc;
    const __bf16* bsrc0 = w2e + (size_t)(d0 + ldr) * FFD + ldc;
    const __bf16* bsrc1 = w2e + (size_t)(d0 + 64 + ldr) * FFD + ldc;
    unsigned aL  = lds_off(As + ldr * 32 + ldc);
    unsigned bL0 = lds_off(Bs + ldr * 32 + ldc);
    unsigned bL1 = lds_off(Bs + (64 + ldr) * 32 + ldc);

    int wv = t >> 5, lane = t & 31;
    int wm = wv >> 2, wn = wv & 3;
    int rit = lane & 15, khalf = lane >> 4;
    v8f acc00 = {}, acc01 = {}, acc10 = {}, acc11 = {};

    const int NSTEP = FFD / 32;
    async_copy_b128(aL, asrc);
    async_copy_b128(bL0, bsrc0);
    async_copy_b128(bL1, bsrc1);

    for (int i = 0; i < NSTEP; ++i) {
        if (i + 1 < NSTEP) {
            int kn = (i + 1) * 32;
            unsigned bo = ((i + 1) & 1) ? 1u : 0u;
            async_copy_b128(aL  + bo * (A_ELEMS * 2), asrc  + kn);
            async_copy_b128(bL0 + bo * (B_ELEMS * 2), bsrc0 + kn);
            async_copy_b128(bL1 + bo * (B_ELEMS * 2), bsrc1 + kn);
            __builtin_prefetch(bsrc0 + kn + 32, 0, 1);
            asm volatile("s_wait_asynccnt 0x3" ::: "memory");
        } else {
            asm volatile("s_wait_asynccnt 0x0" ::: "memory");
        }
        __syncthreads();
        const __bf16* Ab = As + (i & 1) * A_ELEMS;
        const __bf16* Bb = Bs + (i & 1) * B_ELEMS;
        v16bf a0 = load_frag(Ab, wm * 32 + rit,      khalf);
        v16bf a1 = load_frag(Ab, wm * 32 + 16 + rit, khalf);
        v16bf b0 = load_frag(Bb, wn * 32 + rit,      khalf);
        v16bf b1 = load_frag(Bb, wn * 32 + 16 + rit, khalf);
        acc00 = __builtin_amdgcn_wmma_f32_16x16x32_bf16(false, a0, false, b0, (short)0, acc00, false, false);
        acc01 = __builtin_amdgcn_wmma_f32_16x16x32_bf16(false, a0, false, b1, (short)0, acc01, false, false);
        acc10 = __builtin_amdgcn_wmma_f32_16x16x32_bf16(false, a1, false, b0, (short)0, acc10, false, false);
        acc11 = __builtin_amdgcn_wmma_f32_16x16x32_bf16(false, a1, false, b1, (short)0, acc11, false, false);
        __syncthreads();
    }

    int n_in = lane & 15;
    int m_hi = (lane >> 4) * 8;
    int gc0 = d0 + wn * 32 + n_in;
#pragma unroll
    for (int r = 0; r < 8; ++r) {
        int ml0 = wm * 32 + m_hi + r;
        int ml1 = ml0 + 16;
        if (m0 + ml0 < cnt) {
            int slot = base + m0 + ml0;
            int tk = tok[slot];
            float w = wgt[slot];
            atomicAdd(&out[(size_t)tk * DIMD + gc0],      w * acc00[r]);
            atomicAdd(&out[(size_t)tk * DIMD + gc0 + 16], w * acc01[r]);
        }
        if (m0 + ml1 < cnt) {
            int slot = base + m0 + ml1;
            int tk = tok[slot];
            float w = wgt[slot];
            atomicAdd(&out[(size_t)tk * DIMD + gc0],      w * acc10[r]);
            atomicAdd(&out[(size_t)tk * DIMD + gc0 + 16], w * acc11[r]);
        }
    }
}

// ---------------- host launcher ----------------
extern "C" void kernel_launch(void* const* d_in, const int* in_sizes, int n_in,
                              void* d_out, int out_size, void* d_ws, size_t ws_size,
                              hipStream_t stream) {
    const float* x  = (const float*)d_in[0];   // [N_TOK, DIMD]
    const float* Wr = (const float*)d_in[1];   // [NEXP, DIMD]
    const float* W1 = (const float*)d_in[2];   // [NEXP, DIMD, FFD]
    const float* W2 = (const float*)d_in[3];   // [NEXP, FFD, DIMD]
    float* out = (float*)d_out;

    char* ws = (char*)d_ws;
    size_t off = 0;
    auto carve = [&](size_t bytes) -> void* {
        void* p = ws + off;
        off += (bytes + 255) & ~(size_t)255;
        return p;
    };
    __bf16* xb   = (__bf16*)carve((size_t)N_TOK * DIMD * 2);        //   8 MB
    __bf16* w1t  = (__bf16*)carve((size_t)NEXP * DIMD * FFD * 2);   //  64 MB  [e][f][d]
    __bf16* w2t  = (__bf16*)carve((size_t)NEXP * FFD * DIMD * 2);   //  64 MB  [e][d][f]
    __bf16* hbuf = (__bf16*)carve((size_t)SLOTS * FFD * 2);         //  64 MB
    int*    topi = (int*)   carve((size_t)SLOTS * 4);
    float*  topw = (float*) carve((size_t)SLOTS * 4);
    int*    tok  = (int*)   carve((size_t)SLOTS * 4);
    float*  wgt  = (float*) carve((size_t)SLOTS * 4);
    int*    cnts = (int*)   carve((size_t)NEXP * 4);
    int*    offs = (int*)   carve((size_t)(NEXP + 1) * 4);

    hipMemsetAsync(d_out, 0, (size_t)out_size * sizeof(float), stream);

    k_cvt_x<<<(N_TOK * DIMD) / (256 * 4), 256, 0, stream>>>(x, xb);
    // W1: in [DIMD][FFD] -> out [FFD][DIMD]
    k_transpose_cvt<<<dim3(FFD / 64, DIMD / 64, NEXP), 256, 0, stream>>>(W1, w1t, DIMD, FFD);
    // W2: in [FFD][DIMD] -> out [DIMD][FFD]
    k_transpose_cvt<<<dim3(DIMD / 64, FFD / 64, NEXP), 256, 0, stream>>>(W2, w2t, FFD, DIMD);
    k_router<<<N_TOK / 4, 128, 0, stream>>>(x, Wr, topi, topw);
    k_scatter<<<1, 256, 0, stream>>>(topi, topw, tok, wgt, cnts, offs);
    k_ffn1<<<dim3(N_TOK / 64, FFD / 128, NEXP), 256, 0, stream>>>(xb, w1t, tok, cnts, offs, hbuf);
    k_ffn2<<<dim3(N_TOK / 64, DIMD / 128, NEXP), 256, 0, stream>>>(hbuf, w2t, tok, wgt, cnts, offs, out);
}
